// MultiHeadAttention_71794673320528
// MI455X (gfx1250) — matrix operand order repaired and checked by reading, not compiled, not run on hardware
//
#include <hip/hip_runtime.h>

typedef __attribute__((ext_vector_type(16))) _Float16 v16h;
typedef __attribute__((ext_vector_type(8)))  float    v8f;
typedef __attribute__((ext_vector_type(4)))  int      v4i;
typedef __attribute__((ext_vector_type(4)))  float    v4f;

// explicit address spaces: 1 = global (LOADcnt/STOREcnt only), 3 = LDS (DScnt only)
#define AS(n) __attribute__((address_space(n)))
typedef AS(1) const v4i    GCv4i;
typedef AS(1) const v4f    GCv4f;
typedef AS(1) const float  GCf32;
typedef AS(1) const int    GCi32;
typedef AS(1) float        Gf32;
typedef AS(3) _Float16     LH;
typedef AS(3) const _Float16 LCH;
typedef AS(3) const v4i    LCv4i;

#define DIM   512
#define NTOK  64
#define HEADS 8
#define HDIM  64
#define XS    520          // row stride (f16) for X/Q/K LDS regions (bank-skewed, 16B-aligned)
#define VTS   72           // row stride (f16) for transposed V (16B-aligned)
#define KSTEPS (DIM / 32)  // 16 k-steps of 32 for the 512-deep GEMMs

union Frag { v16h h; v4i i[2]; };

static __device__ inline v8f zero8() {
  v8f z = {0.f, 0.f, 0.f, 0.f, 0.f, 0.f, 0.f, 0.f};
  return z;
}

// A-matrix fragment (16x32 f16) from row-major LDS tile at `t` (row stride `stride`).
// lane<16: M=lane, K 0..15 ; lane>=16: M=lane-16, K 16..31.
static __device__ inline v16h loadA_lds(LCH* t, int stride, int lane) {
  const int m = lane & 15, hi = lane >> 4;
  LCH* p = t + m * stride + 16 * hi;
  Frag f;
  f.i[0] = *(LCv4i*)(p);
  f.i[1] = *(LCv4i*)(p + 8);
  return f.h;
}

// B-matrix fragment (32x16 f16) where source is laid out [n][k] row-major (k contiguous).
// lane<16: N=lane, K 0..15 ; lane>=16: N=lane-16, K 16..31.
static __device__ inline v16h loadB_lds(LCH* t, int stride, int lane) {
  const int n = lane & 15, hi = lane >> 4;
  LCH* p = t + n * stride + 16 * hi;
  Frag f;
  f.i[0] = *(LCv4i*)(p);
  f.i[1] = *(LCv4i*)(p + 8);
  return f.h;
}

// B fragment from pre-packed weights: fragment (nt,kk) = 64 consecutive v4i,
// lane's 8 dwords at lane*2. One clause of two global_load_b128.
static __device__ inline v16h loadB_pk(GCv4i* w, int nt, int kk, int lane) {
  GCv4i* p = w + ((nt * KSTEPS + kk) * 64 + lane * 2);
  Frag f;
  f.i[0] = p[0];
  f.i[1] = p[1];
  return f.h;
}

static __device__ inline v8f wmma16(v16h a, v16h b, v8f c) {
  return __builtin_amdgcn_wmma_f32_16x16x32_f16(false, a, false, b, (short)0, c, false, false);
}

// fp32 64x512 global -> f16 LDS (row stride XS), 256 threads. Inputs are
// streaming (read exactly once per launch) -> non-temporal hint keeps the
// caches free for the reused packed weights.
static __device__ inline void load_x(GCf32* src, LH* X, int tid) {
  GCv4f* s4 = (GCv4f*)src;
  for (int i = tid; i < NTOK * DIM / 4; i += 256) {
    const int row = i >> 7;          // 128 float4 per row
    const int c4  = (i & 127) * 4;
    const v4f f = __builtin_nontemporal_load(s4 + i);
    LH* d = X + row * XS + c4;
    d[0] = (_Float16)f[0]; d[1] = (_Float16)f[1];
    d[2] = (_Float16)f[2]; d[3] = (_Float16)f[3];
  }
}

// OUT(64x512) = X(64x512) @ W^T + bias, scaled; store f16 to LDS.
// 4x4 register blocking: each A fragment read from LDS exactly once per k-step,
// 16 independent WMMAs per k-step for latency hiding.
template <bool TRANSPOSED>
static __device__ inline void proj_lds(LCH* Xin, GCv4i* wpk, GCf32* bias, float scale,
                                       LH* Out, int wave, int lane) {
  const int hi = lane >> 4;
  v8f acc[4][4];
#pragma unroll
  for (int i = 0; i < 4; ++i)
#pragma unroll
    for (int rt = 0; rt < 4; ++rt) acc[i][rt] = zero8();

  for (int kk = 0; kk < KSTEPS; ++kk) {
    v16h a[4];
#pragma unroll
    for (int rt = 0; rt < 4; ++rt)
      a[rt] = loadA_lds(Xin + rt * 16 * XS + kk * 32, XS, lane);
#pragma unroll
    for (int i = 0; i < 4; ++i) {
      const v16h b = loadB_pk(wpk, wave + 8 * i, kk, lane);
#pragma unroll
      for (int rt = 0; rt < 4; ++rt)
        acc[i][rt] = wmma16(a[rt], b, acc[i][rt]);
    }
  }

#pragma unroll
  for (int i = 0; i < 4; ++i) {
    const int n = (wave + 8 * i) * 16 + (lane & 15);
    const float bv = bias[n];
#pragma unroll
    for (int rt = 0; rt < 4; ++rt)
#pragma unroll
      for (int r = 0; r < 8; ++r) {
        const float val = (acc[i][rt][r] + bv) * scale;
        const int m = rt * 16 + r + 8 * hi;
        if (TRANSPOSED) Out[n * VTS + m] = (_Float16)val;   // [dim][token]
        else            Out[m * XS + n]  = (_Float16)val;   // [token][dim]
      }
  }
}

// out(64x512 fp32, global) = Xin @ W^T + bias
static __device__ inline void proj_out(LCH* Xin, GCv4i* wpk, GCf32* bias,
                                       Gf32* out, int wave, int lane) {
  const int hi = lane >> 4;
  v8f acc[4][4];
#pragma unroll
  for (int i = 0; i < 4; ++i)
#pragma unroll
    for (int rt = 0; rt < 4; ++rt) acc[i][rt] = zero8();

  for (int kk = 0; kk < KSTEPS; ++kk) {
    v16h a[4];
#pragma unroll
    for (int rt = 0; rt < 4; ++rt)
      a[rt] = loadA_lds(Xin + rt * 16 * XS + kk * 32, XS, lane);
#pragma unroll
    for (int i = 0; i < 4; ++i) {
      const v16h b = loadB_pk(wpk, wave + 8 * i, kk, lane);
#pragma unroll
      for (int rt = 0; rt < 4; ++rt)
        acc[i][rt] = wmma16(a[rt], b, acc[i][rt]);
    }
  }

#pragma unroll
  for (int i = 0; i < 4; ++i) {
    const int n = (wave + 8 * i) * 16 + (lane & 15);
    const float bv = bias[n];
#pragma unroll
    for (int rt = 0; rt < 4; ++rt)
#pragma unroll
      for (int r = 0; r < 8; ++r) {
        const int m = rt * 16 + r + 8 * hi;
        out[(size_t)m * DIM + n] = acc[i][rt][r] + bv;
      }
  }
}

__global__ __launch_bounds__(256) void mha_fused(
    const float* __restrict__ q, const float* __restrict__ k, const float* __restrict__ v,
    const int* __restrict__ mask,
    const float* __restrict__ bq, const float* __restrict__ bk,
    const float* __restrict__ bvv, const float* __restrict__ bp,
    const v4i* __restrict__ wq, const v4i* __restrict__ wk,
    const v4i* __restrict__ wv, const v4i* __restrict__ wp,
    float* __restrict__ out) {
  extern __shared__ _Float16 smem[];
  LH* Xb = (LH*)smem;                  // 64 x XS  (staged input)
  LH* Qb = Xb + NTOK * XS;             // 64 x XS  (Q; later P, per-head columns)
  LH* Kb = Qb + NTOK * XS;             // 64 x XS  (K; later attention output)
  LH* VT = Kb + NTOK * XS;             // 512 x VTS (V transposed: [dim][token])

  GCv4i* wqg = (GCv4i*)wq;
  GCv4i* wkg = (GCv4i*)wk;
  GCv4i* wvg = (GCv4i*)wv;
  GCv4i* wpg = (GCv4i*)wp;
  GCi32* maskg = (GCi32*)mask;

  const int bidx = blockIdx.x;
  const int tid  = threadIdx.x;
  const int wave = tid >> 5, lane = tid & 31;
  const int hi = lane >> 4, ln = lane & 15;
  const size_t base = (size_t)bidx * NTOK * DIM;

  // ---- projections: Q (scaled), K, V(transposed) ----
  load_x((GCf32*)(q + base), Xb, tid);
  __syncthreads();
  proj_lds<false>(Xb, wqg, (GCf32*)bq, 0.125f, Qb, wave, lane);  // SCALE = HEAD_DIM^-0.5
  __syncthreads();
  load_x((GCf32*)(k + base), Xb, tid);
  __syncthreads();
  proj_lds<false>(Xb, wkg, (GCf32*)bk, 1.0f, Kb, wave, lane);
  __syncthreads();
  load_x((GCf32*)(v + base), Xb, tid);
  __syncthreads();
  proj_lds<true>(Xb, wvg, (GCf32*)bvv, 1.0f, VT, wave, lane);
  __syncthreads();

  // ---- attention: one wave per head ----
  const int h = wave;
  bool masked[4];
#pragma unroll
  for (int ct = 0; ct < 4; ++ct)
    masked[ct] = (maskg[bidx * NTOK + ct * 16 + ln] == 0);

  LCH* Qh = Qb + h * HDIM;
  LCH* Kh = Kb + h * HDIM;
  LH*  Ph = Qb + h * HDIM;   // P overlays Q (per-head columns, row-sequenced)

  // hoist all K^T fragments (head-dim 64 = 2 k-steps x 4 col tiles)
  v16h bK[2][4];
#pragma unroll
  for (int kk = 0; kk < 2; ++kk)
#pragma unroll
    for (int ct = 0; ct < 4; ++ct)
      bK[kk][ct] = loadB_lds(Kh + ct * 16 * XS + kk * 32, XS, lane);

  for (int rt = 0; rt < 4; ++rt) {
    v8f s[4] = {zero8(), zero8(), zero8(), zero8()};
#pragma unroll
    for (int kk = 0; kk < 2; ++kk) {
      const v16h aq = loadA_lds(Qh + rt * 16 * XS + kk * 32, XS, lane);
#pragma unroll
      for (int ct = 0; ct < 4; ++ct)
        s[ct] = wmma16(aq, bK[kk][ct], s[ct]);
    }
    // mask + row softmax (a row lives within a 16-lane half -> xor-shuffle reductions)
#pragma unroll
    for (int r = 0; r < 8; ++r) {
      float vals[4];
      float rmax = -3.0e38f;
#pragma unroll
      for (int ct = 0; ct < 4; ++ct) {
        const float x = masked[ct] ? -10000.0f : s[ct][r];
        vals[ct] = x;
        rmax = fmaxf(rmax, x);
      }
#pragma unroll
      for (int d = 1; d < 16; d <<= 1)
        rmax = fmaxf(rmax, __shfl_xor(rmax, d, 32));
      float rsum = 0.f;
#pragma unroll
      for (int ct = 0; ct < 4; ++ct) {
        const float e = __expf(vals[ct] - rmax);
        vals[ct] = e;
        rsum += e;
      }
#pragma unroll
      for (int d = 1; d < 16; d <<= 1)
        rsum += __shfl_xor(rsum, d, 32);
      const float inv = 1.0f / rsum;
      const int row = rt * 16 + r + 8 * hi;
#pragma unroll
      for (int ct = 0; ct < 4; ++ct)
        Ph[row * XS + ct * 16 + ln] = (_Float16)(vals[ct] * inv);
    }
  }

  // ---- P @ V  (attention output overlays K region, per-head columns) ----
  LH* Oh = Kb + h * HDIM;
  v16h bV[2][4];
#pragma unroll
  for (int kk = 0; kk < 2; ++kk)
#pragma unroll
    for (int dt = 0; dt < 4; ++dt)
      bV[kk][dt] = loadB_lds((LCH*)VT + (h * HDIM + dt * 16) * VTS + kk * 32, VTS, lane);

  for (int rt = 0; rt < 4; ++rt) {
    v8f xo[4] = {zero8(), zero8(), zero8(), zero8()};
#pragma unroll
    for (int kk = 0; kk < 2; ++kk) {
      const v16h ap = loadA_lds((LCH*)Ph + rt * 16 * XS + kk * 32, XS, lane);
#pragma unroll
      for (int dt = 0; dt < 4; ++dt)
        xo[dt] = wmma16(ap, bV[kk][dt], xo[dt]);
    }
#pragma unroll
    for (int dt = 0; dt < 4; ++dt)
#pragma unroll
      for (int r = 0; r < 8; ++r) {
        const int row = rt * 16 + r + 8 * hi;
        Oh[row * XS + dt * 16 + ln] = (_Float16)xo[dt][r];
      }
  }
  __syncthreads();

  // ---- output projection to global fp32 ----
  proj_out((LCH*)Kb, wpg, (GCf32*)bp, (Gf32*)(out + base), wave, lane);
}

// Pack fp32 weight [512][512] (row-major W[n][k]) into this file's contiguous-k WMMA B-fragment
// layout as f16: dword index = ((nTile*16 + kStep)*32 + lane)*8 + j holds
// {W[n][k], W[n][k+1]} with n = 16*nTile + (lane&15), k = 32*kStep + 16*(lane>>4) + 2j.
__global__ __launch_bounds__(256) void pack_w(const float* __restrict__ W,
                                              _Float16* __restrict__ P) {
  const int idx = blockIdx.x * 256 + threadIdx.x;   // one f16 pair each; 131072 total
  if (idx >= (DIM * DIM) / 2) return;
  const int j     = idx & 7;
  const int lane  = (idx >> 3) & 31;
  const int kStep = (idx >> 8) & 15;
  const int nTile = idx >> 12;
  const int n = nTile * 16 + (lane & 15);
  const int kk = kStep * 32 + (lane >> 4) * 16 + j * 2;
  GCf32* Wg = (GCf32*)W;
  AS(1) _Float16* Pg = (AS(1) _Float16*)P;
  Pg[2 * idx]     = (_Float16)Wg[n * DIM + kk];
  Pg[2 * idx + 1] = (_Float16)Wg[n * DIM + kk + 1];
}

extern "C" void kernel_launch(void* const* d_in, const int* in_sizes, int n_in,
                              void* d_out, int out_size, void* d_ws, size_t ws_size,
                              hipStream_t stream) {
  const float* q    = (const float*)d_in[0];
  const float* k    = (const float*)d_in[1];
  const float* v    = (const float*)d_in[2];
  const int*   mask = (const int*)d_in[3];
  const float* Wq   = (const float*)d_in[4];
  const float* bq   = (const float*)d_in[5];
  const float* Wk   = (const float*)d_in[6];
  const float* bk   = (const float*)d_in[7];
  const float* Wv   = (const float*)d_in[8];
  const float* bv   = (const float*)d_in[9];
  const float* Wp   = (const float*)d_in[10];
  const float* bp   = (const float*)d_in[11];

  const int B = in_sizes[0] / (NTOK * DIM);   // 2048
  const int WELEM = DIM * DIM;                // f16 elements per packed weight

  _Float16* wpk = (_Float16*)d_ws;            // needs 4 * 512KB = 2MB workspace
  pack_w<<<(WELEM / 2 + 255) / 256, 256, 0, stream>>>(Wq, wpk);
  pack_w<<<(WELEM / 2 + 255) / 256, 256, 0, stream>>>(Wk, wpk + WELEM);
  pack_w<<<(WELEM / 2 + 255) / 256, 256, 0, stream>>>(Wv, wpk + 2 * WELEM);
  pack_w<<<(WELEM / 2 + 255) / 256, 256, 0, stream>>>(Wp, wpk + 3 * WELEM);

  const size_t smem = (size_t)(3 * NTOK * XS + DIM * VTS) * sizeof(_Float16); // 273,408 B
  mha_fused<<<B, 256, smem, stream>>>(
      q, k, v, mask, bq, bk, bv, bp,
      (const v4i*)wpk, (const v4i*)(wpk + WELEM),
      (const v4i*)(wpk + 2 * WELEM), (const v4i*)(wpk + 3 * WELEM),
      (float*)d_out);
}
